// RecursiveMamba1_PrefixScratchpad_635655159993
// MI455X (gfx1250) — matrix operand order repaired and checked by reading, not compiled, not run on hardware
//
#include <hip/hip_runtime.h>
#include <hip/hip_bf16.h>
#include <stdint.h>

// ---------------- model dims ----------------
#define DM     2048            // D_MODEL
#define DI     4096            // D_INNER
#define DS     16              // D_STATE
#define DCONV  4
#define DTR    128             // DT_RANK
#define XDC    (DTR + 2*DS)    // 160 columns of x_dbl
#define MPRE   8               // M prefix
#define BR     64              // BRIDGE
#define NVOCAB 50280
#define NLOOPS 4
#define BB     2
#define TT     1024
#define LL     (MPRE + TT)     // 1032
#define ROWS   (BB * LL)       // 2064
#define OROWS  (BB * TT)       // 2048

typedef __bf16 bf16;
typedef __attribute__((ext_vector_type(16))) __bf16 v16bf;
typedef __attribute__((ext_vector_type(8)))  __bf16 v8bf;
typedef __attribute__((ext_vector_type(8)))  float  v8f;

// =====================================================================
// WMMA GEMM:  C[M x N] = A[M x K] * W[N x K]^T  (row-major, bf16 in, f32 out)
//
// Block = 256 threads = 8 waves. Block tile = 256 rows x 64 cols.
// Each wave: 32x64 (2 M-tiles x 4 N-tiles, 8 accumulators = 64 VGPRs),
// so every B fragment staged in LDS feeds 2 WMMAs.
//
// B panel (64 cols x 32 K, bf16) is staged cooperatively into LDS via
// CDNA5 async loads (global_load_async_to_lds_b128: 256 thr x 16B = 4KB
// panel per K step), double buffered, synchronized with s_wait_asynccnt
// + workgroup barriers. LDS row stride padded to 40 elements (80B) so the
// 16-lane fragment reads hit 16 distinct banks.
//
// A rows are wave-private -> loaded direct from global with prefetch.
// K must be a multiple of 32. M/N tails: loads are index-clamped, stores
// guarded (no EXEC divergence in the K loop, no early returns -> barriers
// are safe).
// =====================================================================
__global__ __launch_bounds__(256)
void wmma_gemm_bf16_kernel(const bf16* __restrict__ A,
                           const bf16* __restrict__ W,
                           float* __restrict__ C,
                           int M, int N, int K)
{
    __shared__ __align__(16) bf16 Bp[2][64 * 40];   // 2 x 5120B panels

    const int tid  = threadIdx.x;
    const int lane = tid & 31;
    const int wave = tid >> 5;
    const int half = lane >> 4;      // lane group 0/1
    const int r    = lane & 15;      // row (A) / col (B) within 16-tile
    const int n0   = blockIdx.x * 64;
    const int m0   = (blockIdx.y * 8 + wave) * 32;   // 2 M-tiles per wave

    // ---- B staging: thread -> (column, 16B chunk of the 32-K row) ----
    const int sn = tid >> 2;                       // 0..63 panel column
    const int sk = tid & 3;                        // 0..3  16B chunk
    const int gn = min(n0 + sn, N - 1);            // clamp OOB columns
    const unsigned long long wbase =
        (unsigned long long)(uintptr_t)(W + (size_t)gn * K + sk * 8);
    const unsigned ldsbase = (unsigned)(uintptr_t)(&Bp[0][0]);
    const unsigned lds0 = ldsbase + (unsigned)(sn * 80 + sk * 16);
    const unsigned lds1 = lds0 + 64 * 40 * 2;

    // ---- A row pointers (clamped for M tail) ----
    const bf16* arow0 = A + (size_t)min(m0 + r,      M - 1) * K;
    const bf16* arow1 = A + (size_t)min(m0 + 16 + r, M - 1) * K;
    const int ofsA0 = half * 8;          // K 0..7   (grp0) / 8..15  (grp1)
    const int ofsA1 = 16 + half * 8;     // K 16..23 (grp0) / 24..31 (grp1)

    v8f acc[8];                          // [mt*4 + nt]
#pragma unroll
    for (int t = 0; t < 8; ++t)
#pragma unroll
        for (int j = 0; j < 8; ++j) acc[t][j] = 0.0f;

    // prologue: stage K step 0 into buffer 0
    asm volatile("global_load_async_to_lds_b128 %0, %1, off"
                 :: "v"(lds0), "v"(wbase) : "memory");

    int cur = 0;
    for (int k0 = 0; k0 < K; k0 += 32) {
        if (k0 + 32 < K) {               // stage next panel into other buffer
            const unsigned long long src = wbase + (unsigned long long)(k0 + 32) * 2;
            const unsigned dst = (cur == 0) ? lds1 : lds0;
            asm volatile("global_load_async_to_lds_b128 %0, %1, off"
                         :: "v"(dst), "v"(src) : "memory");
            asm volatile("s_wait_asynccnt 0x1" ::: "memory");  // own prev done
        } else {
            asm volatile("s_wait_asynccnt 0x0" ::: "memory");
        }
        __syncthreads();                 // whole panel [cur] staged

        // ---- A fragments (ISA Sec 7.12.2 16-bit A layout) ----
        v8bf a0lo = *(const v8bf*)(arow0 + k0 + ofsA0);
        v8bf a0hi = *(const v8bf*)(arow0 + k0 + ofsA1);
        v8bf a1lo = *(const v8bf*)(arow1 + k0 + ofsA0);
        v8bf a1hi = *(const v8bf*)(arow1 + k0 + ofsA1);
        __builtin_prefetch((const void*)(arow0 + k0 + 128), 0, 3);
        __builtin_prefetch((const void*)(arow1 + k0 + 128), 0, 3);
        v16bf afrag0, afrag1;
#pragma unroll
        for (int i = 0; i < 8; ++i) {
            afrag0[i] = a0lo[i]; afrag0[8 + i] = a0hi[i];
            afrag1[i] = a1lo[i]; afrag1[8 + i] = a1hi[i];
        }

        const bf16* bpanel = &Bp[cur][0];
#pragma unroll
        for (int t = 0; t < 4; ++t) {
            // B fragment from LDS: lane = column, grp0 K0-7,16-23, grp1 K8-15,24-31
            const bf16* bp = bpanel + (t * 16 + r) * 40 + half * 8;
            v8bf blo = *(const v8bf*)bp;
            v8bf bhi = *(const v8bf*)(bp + 16);
            v16bf bfrag;
#pragma unroll
            for (int i = 0; i < 8; ++i) { bfrag[i] = blo[i]; bfrag[8 + i] = bhi[i]; }
            acc[t]     = __builtin_amdgcn_wmma_f32_16x16x32_bf16(
                false, afrag0, false, bfrag, (short)0, acc[t],     false, false);
            acc[4 + t] = __builtin_amdgcn_wmma_f32_16x16x32_bf16(
                false, afrag1, false, bfrag, (short)0, acc[4 + t], false, false);
        }
        __syncthreads();                 // all reads done before re-staging
        cur ^= 1;
    }

    // ---- C store: VGPR j -> row mbase + j, col = r (guarded tails) ----
#pragma unroll
    for (int mt = 0; mt < 2; ++mt) {
        const int mbase = m0 + mt * 16 + half * 8;
#pragma unroll
        for (int t = 0; t < 4; ++t) {
            const int n = n0 + t * 16 + r;
            if (n < N) {
#pragma unroll
                for (int j = 0; j < 8; ++j) {
                    const int m = mbase + j;
                    if (m < M) C[(size_t)m * N + n] = acc[mt * 4 + t][j];
                }
            }
        }
    }
}

// =====================================================================
// Elementwise / setup kernels
// =====================================================================
__global__ void to_bf16_kernel(const float* __restrict__ in, bf16* __restrict__ out, size_t n)
{
    size_t i = (size_t)blockIdx.x * blockDim.x + threadIdx.x;
    size_t stride = (size_t)gridDim.x * blockDim.x;
    for (; i < n; i += stride) out[i] = (bf16)in[i];
}

__global__ void aneg_kernel(const float* __restrict__ A_log, float* __restrict__ A_neg)
{
    int i = blockIdx.x * blockDim.x + threadIdx.x;
    if (i < DI * DS) A_neg[i] = -__expf(A_log[i]);
}

__global__ void init_xext_kernel(const int* __restrict__ ids,
                                 const float* __restrict__ emb,
                                 const float* __restrict__ latent,
                                 float* __restrict__ x_ext,
                                 float* __restrict__ x_prompt)
{
    size_t i = (size_t)blockIdx.x * blockDim.x + threadIdx.x;
    if (i >= (size_t)ROWS * DM) return;
    const int d   = (int)(i % DM);
    const int row = (int)(i / DM);
    const int b = row / LL, l = row % LL;
    float v;
    if (l < MPRE) {
        v = latent[l * DM + d];
    } else {
        const int t = l - MPRE;
        const int tok = ids[b * TT + t];
        v = emb[(size_t)tok * DM + d];
        x_prompt[((size_t)b * TT + t) * DM + d] = v;
    }
    x_ext[i] = v;
}

// lifeline add (in place on x_ext) + rope(loop_i) + rmsnorm -> bf16 xn
__global__ __launch_bounds__(256)
void preproc_kernel(float* __restrict__ x_ext,
                    const float* __restrict__ x_prompt,
                    const float* __restrict__ gatev,
                    const float* __restrict__ w,
                    bf16* __restrict__ xn, int loop_i)
{
    __shared__ float red[256];
    const int row = blockIdx.x;                 // 0..ROWS-1
    const int b = row / LL, l = row % LL;
    float* xr = x_ext + (size_t)row * DM;
    const float* xpr = (l >= MPRE)
        ? (x_prompt + ((size_t)b * TT + (l - MPRE)) * DM) : nullptr;

    const int base = threadIdx.x * 8;           // 256 thr * 8 = 2048
    float vals[8];
    float ss = 0.0f;
#pragma unroll
    for (int p = 0; p < 4; ++p) {
        const int d0 = base + 2 * p;
        float x0 = xr[d0], x1 = xr[d0 + 1];
        if (xpr) {
            x0 += gatev[d0]     * xpr[d0];
            x1 += gatev[d0 + 1] * xpr[d0 + 1];
            xr[d0] = x0; xr[d0 + 1] = x1;       // residual base includes lifeline
        }
        const int j = d0 >> 1;                  // pair index
        const float f = (float)loop_i * __expf(-(2.0f * j / (float)DM) * __logf(10000.0f));
        const float c = __cosf(f), s = __sinf(f);
        const float r0 = x0 * c - x1 * s;
        const float r1 = x1 * c + x0 * s;
        vals[2 * p] = r0; vals[2 * p + 1] = r1;
        ss += r0 * r0 + r1 * r1;
    }
    red[threadIdx.x] = ss;
    __syncthreads();
    for (int o = 128; o > 0; o >>= 1) {
        if ((int)threadIdx.x < o) red[threadIdx.x] += red[threadIdx.x + o];
        __syncthreads();
    }
    const float inv = rsqrtf(red[0] / (float)DM + 1e-6f);
    bf16* xnr = xn + (size_t)row * DM;
#pragma unroll
    for (int e = 0; e < 8; ++e) {
        const int d = base + e;
        xnr[d] = (bf16)(vals[e] * inv * w[d]);
    }
}

// causal depthwise conv (width 4) + bias + SiLU; writes f32 (scan) and bf16 (GEMM)
__global__ void conv_silu_kernel(const float* __restrict__ uz,
                                 const float* __restrict__ cw,
                                 const float* __restrict__ cb,
                                 float* __restrict__ u_f,
                                 bf16*  __restrict__ u_b)
{
    size_t i = (size_t)blockIdx.x * blockDim.x + threadIdx.x;
    if (i >= (size_t)ROWS * DI) return;
    const int d   = (int)(i % DI);
    const int row = (int)(i / DI);
    const int b = row / LL, l = row % LL;
    float acc = cb[d];
#pragma unroll
    for (int t = 0; t < DCONV; ++t) {
        const int ls = l - (DCONV - 1) + t;
        if (ls >= 0)
            acc += cw[d * DCONV + t] * uz[((size_t)(b * LL + ls)) * (2 * DI) + d];
    }
    const float s = acc / (1.0f + __expf(-acc));
    u_f[i] = s;
    u_b[i] = (bf16)s;
}

__global__ void extract_dt_kernel(const float* __restrict__ xdbl, bf16* __restrict__ dtA)
{
    size_t i = (size_t)blockIdx.x * blockDim.x + threadIdx.x;
    if (i >= (size_t)ROWS * DTR) return;
    const int row = (int)(i / DTR), c = (int)(i % DTR);
    dtA[i] = (bf16)xdbl[(size_t)row * XDC + c];
}

__global__ void softplus_kernel(float* __restrict__ dt, const float* __restrict__ bias)
{
    size_t i = (size_t)blockIdx.x * blockDim.x + threadIdx.x;
    if (i >= (size_t)ROWS * DI) return;
    const int d = (int)(i % DI);
    const float x = dt[i] + bias[d];
    dt[i] = fmaxf(x, 0.0f) + log1pf(__expf(-fabsf(x)));  // stable softplus
}

// selective scan: lane = state index n (16 per channel), sequential over L
__global__ __launch_bounds__(256)
void scan_kernel(const float* __restrict__ dt,
                 const float* __restrict__ u,
                 const float* __restrict__ xdbl,
                 const float* __restrict__ A_neg,
                 float* __restrict__ y)
{
    const int gid  = blockIdx.x * blockDim.x + threadIdx.x;
    const int n    = gid & 15;
    const int pair = gid >> 4;                 // (b, d)
    if (pair >= BB * DI) return;               // exact multiple -> never diverges
    const int b = pair / DI, d = pair % DI;
    const float a = A_neg[d * DS + n];
    float h = 0.0f;
    const size_t rb = (size_t)b * LL;
    for (int l = 0; l < LL; ++l) {
        const size_t row = rb + l;
        const float dtv = dt[row * DI + d];    // broadcast across the 16 lanes
        const float uv  = u[row * DI + d];
        const float Bn  = xdbl[row * XDC + DTR + n];
        const float Cn  = xdbl[row * XDC + DTR + DS + n];
        h = h * __expf(dtv * a) + dtv * uv * Bn;
        float p = h * Cn;
        p += __shfl_xor(p, 1, 32);
        p += __shfl_xor(p, 2, 32);
        p += __shfl_xor(p, 4, 32);
        p += __shfl_xor(p, 8, 32);
        if (n == 0) y[row * DI + d] = p;
    }
}

__global__ void gate_kernel(const float* __restrict__ y,
                            const float* __restrict__ u,
                            const float* __restrict__ uz,
                            const float* __restrict__ Dsk,
                            bf16* __restrict__ yg)
{
    size_t i = (size_t)blockIdx.x * blockDim.x + threadIdx.x;
    if (i >= (size_t)ROWS * DI) return;
    const int d   = (int)(i % DI);
    const int row = (int)(i / DI);
    const float z  = uz[(size_t)row * (2 * DI) + DI + d];
    const float sz = z / (1.0f + __expf(-z));
    yg[i] = (bf16)((y[i] + u[i] * Dsk[d]) * sz);
}

__global__ void add_kernel(float* __restrict__ dst, const float* __restrict__ src, size_t n)
{
    size_t i = (size_t)blockIdx.x * blockDim.x + threadIdx.x;
    size_t stride = (size_t)gridDim.x * blockDim.x;
    for (; i < n; i += stride) dst[i] += src[i];
}

__global__ __launch_bounds__(256)
void final_norm_kernel(const float* __restrict__ x_ext,
                       const float* __restrict__ w,
                       bf16* __restrict__ xout)
{
    __shared__ float red[256];
    const int orow = blockIdx.x;               // 0..OROWS-1
    const int b = orow / TT, t = orow % TT;
    const float* xr = x_ext + ((size_t)(b * LL + MPRE + t)) * DM;
    const int base = threadIdx.x * 8;
    float vals[8];
    float ss = 0.0f;
#pragma unroll
    for (int e = 0; e < 8; ++e) { vals[e] = xr[base + e]; ss += vals[e] * vals[e]; }
    red[threadIdx.x] = ss;
    __syncthreads();
    for (int o = 128; o > 0; o >>= 1) {
        if ((int)threadIdx.x < o) red[threadIdx.x] += red[threadIdx.x + o];
        __syncthreads();
    }
    const float inv = rsqrtf(red[0] / (float)DM + 1e-6f);
    bf16* xo = xout + (size_t)orow * DM;
#pragma unroll
    for (int e = 0; e < 8; ++e)
        xo[base + e] = (bf16)(vals[e] * inv * w[base + e]);
}

// =====================================================================
// Host side
// =====================================================================
static inline size_t alignup256(size_t x) { return (x + 255) & ~(size_t)255; }
static inline unsigned gblocks(size_t n)  { return (unsigned)((n + 255) / 256); }
static inline unsigned gblocks_cap(size_t n)
{
    size_t g = (n + 255) / 256;
    return (unsigned)(g > 32768 ? 32768 : g);
}

static void launch_gemm(const bf16* A, const bf16* W, float* C,
                        int M, int N, int K, hipStream_t s)
{
    dim3 grid((unsigned)((N + 63) / 64), (unsigned)((M + 255) / 256));
    wmma_gemm_bf16_kernel<<<grid, 256, 0, s>>>(A, W, C, M, N, K);
}

extern "C" void kernel_launch(void* const* d_in, const int* in_sizes, int n_in,
                              void* d_out, int out_size, void* d_ws, size_t ws_size,
                              hipStream_t stream)
{
    (void)in_sizes; (void)n_in; (void)out_size; (void)ws_size;
    // inputs (setup_inputs order)
    const int*   ids       = (const int*)  d_in[0];
    const float* emb       = (const float*)d_in[1];
    const float* latent    = (const float*)d_in[2];
    const float* lifeline  = (const float*)d_in[3];
    const float* loopnw    = (const float*)d_in[4];
    const float* in_proj_w = (const float*)d_in[5];
    const float* conv_w    = (const float*)d_in[6];
    const float* conv_b    = (const float*)d_in[7];
    const float* x_proj_w  = (const float*)d_in[8];
    const float* dt_proj_w = (const float*)d_in[9];
    const float* dt_proj_b = (const float*)d_in[10];
    const float* A_log     = (const float*)d_in[11];
    const float* D_skip    = (const float*)d_in[12];
    const float* out_proj_w= (const float*)d_in[13];
    const float* bdown_w   = (const float*)d_in[14];
    const float* bup_w     = (const float*)d_in[15];
    const float* norm_f_w  = (const float*)d_in[16];
    const float* lm_head_w = (const float*)d_in[17];
    float* out = (float*)d_out;

    // -------- workspace layout --------
    char* base = (char*)d_ws;
    size_t off = 0;
    auto take = [&](size_t bytes) -> char* {
        char* p = base + off; off = alignup256(off + bytes); return p;
    };
    float* x_prompt = (float*)take((size_t)OROWS * DM * 4);
    float* x_ext    = (float*)take((size_t)ROWS  * DM * 4);
    bf16*  xn       = (bf16*) take((size_t)ROWS  * DM * 2);    // also reused for bridge input
    bf16*  w_in     = (bf16*) take((size_t)2 * DI * DM * 2);
    bf16*  w_xp     = (bf16*) take((size_t)XDC * DI * 2);
    bf16*  w_dt     = (bf16*) take((size_t)DI * DTR * 2);
    bf16*  w_out    = (bf16*) take((size_t)DM * DI * 2);
    bf16*  w_bd     = (bf16*) take((size_t)BR * DM * 2);
    bf16*  w_bu     = (bf16*) take((size_t)DM * BR * 2);
    float* A_neg    = (float*)take((size_t)DI * DS * 4);
    float* xdbl     = (float*)take((size_t)ROWS * XDC * 4);
    bf16*  dtA      = (bf16*) take((size_t)ROWS * DTR * 2);
    float* br1      = (float*)take((size_t)ROWS * BR * 4);
    bf16*  br1_b    = (bf16*) take((size_t)ROWS * BR * 2);
    float* tmp      = (float*)take((size_t)ROWS * DM * 4);
    bf16*  xout     = (bf16*) take((size_t)OROWS * DM * 2);
    // scratch region: per-loop buffers alias lm_head bf16 weights (dead after loops)
    const size_t scratch0 = off;
    float* uz   = (float*)take((size_t)ROWS * 2 * DI * 4);
    float* u_f  = (float*)take((size_t)ROWS * DI * 4);
    bf16*  u_b  = (bf16*) take((size_t)ROWS * DI * 2);
    float* dtb  = (float*)take((size_t)ROWS * DI * 4);
    float* yv   = (float*)take((size_t)ROWS * DI * 4);
    bf16*  yg   = (bf16*) take((size_t)ROWS * DI * 2);
    bf16*  w_lm = (bf16*)(base + scratch0);   // (size_t)NVOCAB*DM*2 exceeds the per-loop scratch buffers by ~3MB

    // -------- weight conversion (stateless per call) --------
    to_bf16_kernel<<<gblocks_cap((size_t)2 * DI * DM), 256, 0, stream>>>(in_proj_w,  w_in,  (size_t)2 * DI * DM);
    to_bf16_kernel<<<gblocks_cap((size_t)XDC * DI),    256, 0, stream>>>(x_proj_w,   w_xp,  (size_t)XDC * DI);
    to_bf16_kernel<<<gblocks_cap((size_t)DI * DTR),    256, 0, stream>>>(dt_proj_w,  w_dt,  (size_t)DI * DTR);
    to_bf16_kernel<<<gblocks_cap((size_t)DM * DI),     256, 0, stream>>>(out_proj_w, w_out, (size_t)DM * DI);
    to_bf16_kernel<<<gblocks_cap((size_t)BR * DM),     256, 0, stream>>>(bdown_w,    w_bd,  (size_t)BR * DM);
    to_bf16_kernel<<<gblocks_cap((size_t)DM * BR),     256, 0, stream>>>(bup_w,      w_bu,  (size_t)DM * BR);
    aneg_kernel<<<gblocks((size_t)DI * DS), 256, 0, stream>>>(A_log, A_neg);

    // -------- embed + prefix concat --------
    init_xext_kernel<<<gblocks((size_t)ROWS * DM), 256, 0, stream>>>(ids, emb, latent, x_ext, x_prompt);

    // -------- N_LOOPS recursive mamba blocks --------
    for (int loop_i = 0; loop_i < NLOOPS; ++loop_i) {
        preproc_kernel<<<ROWS, 256, 0, stream>>>(x_ext, x_prompt, lifeline, loopnw, xn, loop_i);
        launch_gemm(xn, w_in, uz, ROWS, 2 * DI, DM, stream);                // u,z = xn @ in_proj^T
        conv_silu_kernel<<<gblocks((size_t)ROWS * DI), 256, 0, stream>>>(uz, conv_w, conv_b, u_f, u_b);
        launch_gemm(u_b, w_xp, xdbl, ROWS, XDC, DI, stream);                // x_dbl = u @ x_proj^T
        extract_dt_kernel<<<gblocks((size_t)ROWS * DTR), 256, 0, stream>>>(xdbl, dtA);
        launch_gemm(dtA, w_dt, dtb, ROWS, DI, DTR, stream);                 // dt @ dt_proj^T
        softplus_kernel<<<gblocks((size_t)ROWS * DI), 256, 0, stream>>>(dtb, dt_proj_b);
        scan_kernel<<<(BB * DI * DS) / 256, 256, 0, stream>>>(dtb, u_f, xdbl, A_neg, yv);
        gate_kernel<<<gblocks((size_t)ROWS * DI), 256, 0, stream>>>(yv, u_f, uz, D_skip, yg);
        launch_gemm(yg, w_out, tmp, ROWS, DM, DI, stream);                  // @ out_proj^T
        add_kernel<<<gblocks_cap((size_t)ROWS * DM), 256, 0, stream>>>(x_ext, tmp, (size_t)ROWS * DM);
    }

    // -------- bridge (low-rank) + residual --------
    to_bf16_kernel<<<gblocks_cap((size_t)ROWS * DM), 256, 0, stream>>>(x_ext, xn, (size_t)ROWS * DM);
    launch_gemm(xn, w_bd, br1, ROWS, BR, DM, stream);
    to_bf16_kernel<<<gblocks_cap((size_t)ROWS * BR), 256, 0, stream>>>(br1, br1_b, (size_t)ROWS * BR);
    launch_gemm(br1_b, w_bu, tmp, ROWS, DM, BR, stream);
    add_kernel<<<gblocks_cap((size_t)ROWS * DM), 256, 0, stream>>>(x_ext, tmp, (size_t)ROWS * DM);

    // -------- final norm + lm head --------
    final_norm_kernel<<<OROWS, 256, 0, stream>>>(x_ext, norm_f_w, xout);
    to_bf16_kernel<<<gblocks_cap((size_t)NVOCAB * DM), 256, 0, stream>>>(lm_head_w, w_lm, (size_t)NVOCAB * DM);
    launch_gemm(xout, w_lm, out, OROWS, NVOCAB, DM, stream);                // logits, N-tail guarded
}
